// _NonLocalBlockND_16080357556479
// MI455X (gfx1250) — matrix operand order repaired and checked by reading, not compiled, not run on hardware
//
#include <hip/hip_runtime.h>
#include <stdint.h>

// ---------------------------------------------------------------------------
// Types for CDNA5 WMMA (gfx1250, wave32)
// ---------------------------------------------------------------------------
typedef __attribute__((ext_vector_type(16))) __bf16 v16bf;
typedef __attribute__((ext_vector_type(4)))  __bf16 v4bf;
typedef __attribute__((ext_vector_type(8)))  float  v8f;

union Frag16 {
    v16bf v;
    uint4 q[2];          // 32 bytes = 16 bf16 halves
};

__device__ __forceinline__ float redmax16(float v) {
    #pragma unroll
    for (int m = 1; m < 16; m <<= 1) v = fmaxf(v, __shfl_xor(v, m, 16));
    return v;
}
__device__ __forceinline__ float redsum16(float v) {
    #pragma unroll
    for (int m = 1; m < 16; m <<= 1) v += __shfl_xor(v, m, 16);
    return v;
}

// Problem constants
#define BATCH 16
#define CIN   256
#define CI    128
#define NPix  4096   // 64*64
#define NKey  1024   // 32*32

// ---------------------------------------------------------------------------
// Kernel 0: cast weights to bf16; W_w transposed to [Ci][C]
// ---------------------------------------------------------------------------
__global__ void prep_kernel(const float* __restrict__ g_w, const float* __restrict__ th_w,
                            const float* __restrict__ ph_w, const float* __restrict__ W_w,
                            __bf16* __restrict__ gbf, __bf16* __restrict__ tbf,
                            __bf16* __restrict__ pbf, __bf16* __restrict__ Wt)
{
    int tid = blockIdx.x * blockDim.x + threadIdx.x;
    if (tid >= 4 * CI * CIN) return;
    int seg = tid >> 15;           // 32768 per segment
    int i   = tid & 32767;
    if      (seg == 0) gbf[i] = (__bf16)g_w[i];
    else if (seg == 1) tbf[i] = (__bf16)th_w[i];
    else if (seg == 2) pbf[i] = (__bf16)ph_w[i];
    else { int ci = i >> 8, c = i & 255; Wt[i] = (__bf16)W_w[c * CI + ci]; } // Wt[ci][c]
}

// ---------------------------------------------------------------------------
// Kernel 1: conv1x1 as bf16 WMMA GEMM:  out[ci][n] = sum_c w[ci][c]*x[b][c][n]+bias
// Block = 256 thr (8 waves); block tile = 128 rows (all Ci) x 128 pixels.
// x tile f32 -> bf16 double-buffered in LDS (float4 loads, bf16x4 stores).
// mode 0: store theta-major [B,N,Ci]; mode 1: store row-major [B,Ci,N].
// ---------------------------------------------------------------------------
__global__ void __launch_bounds__(256) conv_kernel(
    const float* __restrict__ x,        // [B,256,4096] f32
    const __bf16* __restrict__ w,       // [128,256] bf16
    const float* __restrict__ bias,     // [128]
    __bf16* __restrict__ out, int mode)
{
    const int b    = blockIdx.y;
    const int n0   = blockIdx.x * 128;
    const int tid  = threadIdx.x;
    const int wave = tid >> 5;
    const int lane = tid & 31;
    const int M    = lane & 15;
    const int half = lane >> 4;
    const int kb   = half ? 8 : 0;
    const int m0   = wave * 16;            // 8 waves cover 128 Ci rows

    __shared__ __align__(16) __bf16 xs[2][32 * 128];

    auto fill = [&](int buf, int k0) {
        #pragma unroll
        for (int i = 0; i < 4; ++i) {               // 256 thr * 4 float4 = 32x128
            int idx4 = (i * 256 + tid) * 4;
            int kk   = idx4 >> 7;
            int nn   = idx4 & 127;
            const float4 f = *reinterpret_cast<const float4*>(
                &x[((size_t)(b * CIN + k0 + kk)) * NPix + n0 + nn]);
            v4bf h;
            h[0] = (__bf16)f.x; h[1] = (__bf16)f.y;
            h[2] = (__bf16)f.z; h[3] = (__bf16)f.w;
            *reinterpret_cast<v4bf*>(&xs[buf][idx4]) = h;
        }
    };

    v8f acc[8];
    #pragma unroll
    for (int j = 0; j < 8; ++j) acc[j] = {};

    fill(0, 0);
    for (int kt = 0; kt < 8; ++kt) {
        __syncthreads();
        if (kt < 7) fill((kt + 1) & 1, (kt + 1) * 32);

        const int k0 = kt * 32;
        Frag16 aW;                                  // A: weights 16x32
        const __bf16* wb = w + (size_t)(m0 + M) * CIN + k0;
        aW.q[0] = *reinterpret_cast<const uint4*>(wb + kb);
        aW.q[1] = *reinterpret_cast<const uint4*>(wb + kb + 16);

        Frag16 bX[8];                               // preload all 8 B frags
        const __bf16* ls = &xs[kt & 1][M];
        #pragma unroll
        for (int j = 0; j < 8; ++j) {
            #pragma unroll
            for (int s = 0; s < 16; ++s)
                bX[j].v[s] = ls[(kb + (s & 7) + (s >> 3) * 16) * 128 + j * 16];
        }
        #pragma unroll
        for (int j = 0; j < 8; ++j)
            acc[j] = __builtin_amdgcn_wmma_f32_16x16x32_bf16(
                false, aW.v, false, bX[j].v, (short)0, acc[j], false, false);
    }

    #pragma unroll
    for (int v = 0; v < 8; ++v) {
        int row  = m0 + v + half * 8;                  // Ci index
        float bv = bias[row];
        #pragma unroll
        for (int j = 0; j < 8; ++j) {
            int   col = n0 + j * 16 + M;               // pixel index
            float val = acc[j][v] + bv;
            if (mode == 0)
                out[((size_t)(b * NPix) + col) * CI + row] = (__bf16)val;   // [B,N,Ci]
            else
                out[((size_t)(b * CI) + row) * NPix + col] = (__bf16)val;   // [B,Ci,N]
        }
    }
}

// ---------------------------------------------------------------------------
// Kernel 2: 2x2 maxpool of a full-res conv ([B,Ci,4096] bf16).
// mode 0 -> [B,Ci,1024] (key-major for phi B-fragments)
// mode 1 -> [B,1024,Ci] (key-row-major for g B-fragments)
// ---------------------------------------------------------------------------
__global__ void pool_kernel(const __bf16* __restrict__ full,
                            __bf16* __restrict__ outp, int mode)
{
    int idx = blockIdx.x * blockDim.x + threadIdx.x;
    if (idx >= BATCH * CI * NKey) return;
    int mp = idx & 1023;
    int ci = (idx >> 10) & 127;
    int b  = idx >> 17;
    int ph = mp >> 5, pw = mp & 31;
    const __bf16* base = full + ((size_t)(b * CI + ci)) * NPix;
    int n00 = (2 * ph) * 64 + 2 * pw;
    float mx = fmaxf(fmaxf((float)base[n00],      (float)base[n00 + 1]),
                     fmaxf((float)base[n00 + 64], (float)base[n00 + 65]));
    if (mode == 0) outp[((size_t)(b * CI   + ci)) * NKey + mp] = (__bf16)mx;
    else           outp[((size_t)(b * NKey + mp)) * CI   + ci] = (__bf16)mx;
}

// ---------------------------------------------------------------------------
// Kernel 3: fused flash-attention.
// Each wave owns 16 queries; streams 32-key tiles; online softmax; O (16x128)
// f32 stays in VGPRs (8 WMMA accumulators). S->P transpose via per-wave LDS.
// ---------------------------------------------------------------------------
__global__ void __launch_bounds__(128) attn_kernel(
    const __bf16* __restrict__ theta, // [B,4096,128]
    const __bf16* __restrict__ phiP,  // [B,128,1024]
    const __bf16* __restrict__ gP,    // [B,1024,128]
    __bf16* __restrict__ y)           // [B,4096,128]
{
    const int b    = blockIdx.y;
    const int wave = threadIdx.x >> 5;
    const int lane = threadIdx.x & 31;
    const int q0   = blockIdx.x * 64 + wave * 16;
    const int M    = lane & 15;
    const int half = lane >> 4;
    const int kb   = half ? 8 : 0;

    __shared__ __align__(16) __bf16 pbuf[4][16 * 32];
    __bf16* pw = pbuf[wave];

    // A fragments for this wave's 16 query rows, K = 0..127 in 4 chunks of 32
    Frag16 aT[4];
    {
        const __bf16* tb = theta + ((size_t)(b * NPix + q0 + M)) * CI;
        #pragma unroll
        for (int kc = 0; kc < 4; ++kc) {
            aT[kc].q[0] = *reinterpret_cast<const uint4*>(tb + kc * 32 + kb);
            aT[kc].q[1] = *reinterpret_cast<const uint4*>(tb + kc * 32 + kb + 16);
        }
    }

    v8f o[8];
    #pragma unroll
    for (int j = 0; j < 8; ++j) o[j] = {};
    float mrow[8], lrow[8];
    #pragma unroll
    for (int v = 0; v < 8; ++v) { mrow[v] = -1e30f; lrow[v] = 0.0f; }

    for (int t = 0; t < NKey / 32; ++t) {
        const int m0 = t * 32;

        // ---- S = theta * phi : preload 8 B frags, then one WMMA chain ----
        Frag16 bp[8];
        #pragma unroll
        for (int kc = 0; kc < 4; ++kc) {
            const __bf16* pb = phiP + ((size_t)(b * CI + kc * 32)) * NKey + m0 + M;
            #pragma unroll
            for (int s = 0; s < 16; ++s) {
                const int kk = kb + (s & 7) + (s >> 3) * 16;
                bp[2 * kc].v[s]     = pb[kk * NKey];
                bp[2 * kc + 1].v[s] = pb[kk * NKey + 16];
            }
        }
        v8f s0 = {}, s1 = {};
        #pragma unroll
        for (int kc = 0; kc < 4; ++kc) {
            s0 = __builtin_amdgcn_wmma_f32_16x16x32_bf16(false, aT[kc].v, false,
                     bp[2 * kc].v,     (short)0, s0, false, false);
            s1 = __builtin_amdgcn_wmma_f32_16x16x32_bf16(false, aT[kc].v, false,
                     bp[2 * kc + 1].v, (short)0, s1, false, false);
        }

        // ---- online softmax across this 32-key tile ----
        #pragma unroll
        for (int v = 0; v < 8; ++v) {
            float tm = redmax16(fmaxf(s0[v], s1[v]));
            float mn = fmaxf(mrow[v], tm);
            float sc = __expf(mrow[v] - mn);
            float p0 = __expf(s0[v] - mn);
            float p1 = __expf(s1[v] - mn);
            float rs = redsum16(p0 + p1);
            lrow[v] = lrow[v] * sc + rs;
            mrow[v] = mn;
            #pragma unroll
            for (int j = 0; j < 8; ++j) o[j][v] *= sc;
            int row = v + half * 8;
            pw[row * 32 + M]      = (__bf16)p0;        // C-layout -> LDS 16x32
            pw[row * 32 + 16 + M] = (__bf16)p1;
        }
        asm volatile("s_wait_dscnt 0" ::: "memory");

        // ---- O += P * g : preload P A-frag + 8 g B-frags, then WMMA chain ----
        Frag16 pA;
        pA.q[0] = *reinterpret_cast<const uint4*>(pw + M * 32 + kb);
        pA.q[1] = *reinterpret_cast<const uint4*>(pw + M * 32 + kb + 16);

        Frag16 bg[8];
        const __bf16* gb = gP + ((size_t)(b * NKey + m0)) * CI + M;
        #pragma unroll
        for (int j = 0; j < 8; ++j) {
            #pragma unroll
            for (int s = 0; s < 16; ++s)
                bg[j].v[s] = gb[(kb + (s & 7) + (s >> 3) * 16) * CI + j * 16];
        }
        #pragma unroll
        for (int j = 0; j < 8; ++j)
            o[j] = __builtin_amdgcn_wmma_f32_16x16x32_bf16(false, pA.v, false, bg[j].v,
                                                           (short)0, o[j], false, false);
    }

    #pragma unroll
    for (int v = 0; v < 8; ++v) {
        float inv = 1.0f / lrow[v];
        int   row = v + half * 8;
        __bf16* yb = y + ((size_t)(b * NPix + q0 + row)) * CI;
        #pragma unroll
        for (int j = 0; j < 8; ++j) yb[j * 16 + M] = (__bf16)(o[j][v] * inv);
    }
}

// ---------------------------------------------------------------------------
// Kernel 4: W conv: Wy[b][c][n] = sum_ci W_w[c][ci]*y[b][n][ci] + W_b[c]
// GEMM A = y tile (queries x ci), B = Wt (ci x c). Output stored [B,C,N] f32.
// ---------------------------------------------------------------------------
__global__ void __launch_bounds__(256) wconv_kernel(
    const __bf16* __restrict__ y,    // [B,4096,128]
    const __bf16* __restrict__ Wt,   // [128,256]
    const float* __restrict__ Wb,    // [256]
    float* __restrict__ Wy)          // [B,256,4096]
{
    const int b    = blockIdx.z;
    const int r0   = blockIdx.x * 128;      // query rows
    const int c0   = blockIdx.y * 128;      // output channels
    const int tid  = threadIdx.x;
    const int wave = tid >> 5;
    const int lane = tid & 31;
    const int M    = lane & 15;
    const int half = lane >> 4;
    const int kb   = half ? 8 : 0;
    const int m0   = r0 + wave * 16;

    v8f acc[8];
    #pragma unroll
    for (int j = 0; j < 8; ++j) acc[j] = {};

    #pragma unroll
    for (int k0 = 0; k0 < CI; k0 += 32) {
        Frag16 aY;
        const __bf16* ybp = y + ((size_t)(b * NPix) + m0 + M) * CI + k0;
        aY.q[0] = *reinterpret_cast<const uint4*>(ybp + kb);
        aY.q[1] = *reinterpret_cast<const uint4*>(ybp + kb + 16);

        Frag16 bW[8];                            // preload all 8 B frags
        const __bf16* wbp = Wt + (size_t)k0 * CIN + c0 + M;
        #pragma unroll
        for (int j = 0; j < 8; ++j) {
            #pragma unroll
            for (int s = 0; s < 16; ++s)
                bW[j].v[s] = wbp[(kb + (s & 7) + (s >> 3) * 16) * CIN + j * 16];
        }
        #pragma unroll
        for (int j = 0; j < 8; ++j)
            acc[j] = __builtin_amdgcn_wmma_f32_16x16x32_bf16(
                false, aY.v, false, bW[j].v, (short)0, acc[j], false, false);
    }
    #pragma unroll
    for (int v = 0; v < 8; ++v) {
        int qrow = m0 + v + half * 8;
        #pragma unroll
        for (int j = 0; j < 8; ++j) {
            int c = c0 + j * 16 + M;
            Wy[((size_t)(b * CIN) + c) * NPix + qrow] = acc[j][v] + Wb[c];
        }
    }
}

// ---------------------------------------------------------------------------
// Kernel 5: BN batch statistics per channel (mean, inv_std over B*N)
// ---------------------------------------------------------------------------
__global__ void __launch_bounds__(256) bnstats_kernel(const float* __restrict__ Wy,
                                                      float* __restrict__ stats)
{
    int c = blockIdx.x, tid = threadIdx.x;
    float s = 0.f, s2 = 0.f;
    for (int b = 0; b < BATCH; ++b) {
        const float* p = Wy + ((size_t)(b * CIN) + c) * NPix;
        for (int n = tid; n < NPix; n += 256) { float v = p[n]; s += v; s2 += v * v; }
    }
    __shared__ float sh[256], sh2[256];
    sh[tid] = s; sh2[tid] = s2; __syncthreads();
    for (int st = 128; st > 0; st >>= 1) {
        if (tid < st) { sh[tid] += sh[tid + st]; sh2[tid] += sh2[tid + st]; }
        __syncthreads();
    }
    if (tid == 0) {
        float Nf   = (float)BATCH * (float)NPix;
        float mean = sh[0] / Nf;
        float var  = sh2[0] / Nf - mean * mean;
        stats[c]       = mean;
        stats[256 + c] = rsqrtf(var + 1e-5f);
    }
}

// ---------------------------------------------------------------------------
// Kernel 6: normalize + affine + residual; also forward inp0
// out layout: [inp0 (1 float)] ++ z [B,C,H,W]
// ---------------------------------------------------------------------------
__global__ void final_kernel(const float* __restrict__ Wy, const float* __restrict__ x,
                             const float* __restrict__ stats,
                             const float* __restrict__ gamma, const float* __restrict__ beta,
                             const float* __restrict__ inp0, float* __restrict__ out)
{
    size_t idx = (size_t)blockIdx.x * blockDim.x + threadIdx.x;
    if (idx == 0) out[0] = inp0[0];
    if (idx >= (size_t)BATCH * CIN * NPix) return;
    int c = (int)((idx >> 12) & 255);
    float z = (Wy[idx] - stats[c]) * stats[256 + c] * gamma[c] + beta[c] + x[idx];
    out[1 + idx] = z;
}

// ---------------------------------------------------------------------------
extern "C" void kernel_launch(void* const* d_in, const int* in_sizes, int n_in,
                              void* d_out, int out_size, void* d_ws, size_t ws_size,
                              hipStream_t stream) {
    (void)in_sizes; (void)n_in; (void)out_size; (void)ws_size;
    const float* inp0  = (const float*)d_in[0];
    const float* x     = (const float*)d_in[1];
    const float* g_w   = (const float*)d_in[2];
    const float* g_b   = (const float*)d_in[3];
    const float* th_w  = (const float*)d_in[4];
    const float* th_b  = (const float*)d_in[5];
    const float* ph_w  = (const float*)d_in[6];
    const float* ph_b  = (const float*)d_in[7];
    const float* W_w   = (const float*)d_in[8];
    const float* W_b   = (const float*)d_in[9];
    const float* gamma = (const float*)d_in[10];
    const float* beta  = (const float*)d_in[11];
    float* out = (float*)d_out;

    char*  ws  = (char*)d_ws;
    size_t off = 0;
    auto take = [&](size_t bytes) -> char* {
        char* p = ws + off;
        off = (off + bytes + 255) & ~(size_t)255;
        return p;
    };
    __bf16* theta  = (__bf16*)take((size_t)BATCH * NPix * CI * 2);
    __bf16* fullcv = (__bf16*)take((size_t)BATCH * CI * NPix * 2);
    __bf16* phiP   = (__bf16*)take((size_t)BATCH * CI * NKey * 2);
    __bf16* gPool  = (__bf16*)take((size_t)BATCH * NKey * CI * 2);
    __bf16* yb     = (__bf16*)take((size_t)BATCH * NPix * CI * 2);
    float*  Wy     = (float*)take((size_t)BATCH * CIN * NPix * 4);
    __bf16* gbf    = (__bf16*)take((size_t)CI * CIN * 2);
    __bf16* tbf    = (__bf16*)take((size_t)CI * CIN * 2);
    __bf16* pbf    = (__bf16*)take((size_t)CI * CIN * 2);
    __bf16* Wt     = (__bf16*)take((size_t)CI * CIN * 2);
    float*  stats  = (float*)take(512 * 4);

    prep_kernel<<<512, 256, 0, stream>>>(g_w, th_w, ph_w, W_w, gbf, tbf, pbf, Wt);

    dim3 cgrid(NPix / 128, BATCH);
    conv_kernel<<<cgrid, 256, 0, stream>>>(x, tbf, th_b, theta, 0);    // theta [B,N,Ci]
    conv_kernel<<<cgrid, 256, 0, stream>>>(x, pbf, ph_b, fullcv, 1);   // phi full-res
    pool_kernel<<<(BATCH * CI * NKey) / 256, 256, 0, stream>>>(fullcv, phiP, 0);
    conv_kernel<<<cgrid, 256, 0, stream>>>(x, gbf, g_b, fullcv, 1);    // g full-res
    pool_kernel<<<(BATCH * CI * NKey) / 256, 256, 0, stream>>>(fullcv, gPool, 1);

    attn_kernel<<<dim3(NPix / 64, BATCH), 128, 0, stream>>>(theta, phiP, gPool, yb);

    wconv_kernel<<<dim3(NPix / 128, CIN / 128, BATCH), 256, 0, stream>>>(yb, Wt, W_b, Wy);
    bnstats_kernel<<<CIN, 256, 0, stream>>>(Wy, stats);
    final_kernel<<<(BATCH * CIN * NPix) / 256, 256, 0, stream>>>(Wy, x, stats, gamma, beta, inp0, out);
}
